// EfficientSpatialWindowAttention_4466765987896
// MI455X (gfx1250) — compile-verified, matrix operand order checked by reading, not run on hardware
//
#include <hip/hip_runtime.h>

// ---------------- problem constants ----------------
// B=16, N=4096 (64x64), C=256, WS=8, NH=8, HD=32, LK=32, G=16
// windows: 8x8 per image -> BW = 1024, tokens/window NW = 64, rows = 65536

typedef __bf16 bf16_t;
typedef __attribute__((ext_vector_type(16))) __bf16 v16bf;
typedef __attribute__((ext_vector_type(8)))  float  v8f;
typedef unsigned int v4u __attribute__((ext_vector_type(4)));
typedef int          v8i __attribute__((ext_vector_type(8)));
typedef int          v4i __attribute__((ext_vector_type(4)));

union FragU { v16bf v; uint4 u[2]; };

__device__ __forceinline__ unsigned short f2bf(float f) {
  unsigned int u = __float_as_uint(f);
  unsigned int r = 0x7fffu + ((u >> 16) & 1u);   // round-to-nearest-even
  return (unsigned short)((u + r) >> 16);
}

// CDNA5 async global->LDS copy (ASYNCcnt-tracked, no VGPR round trip).
// GV mode: per-lane 64-bit global address, per-lane 32-bit LDS byte address.
__device__ __forceinline__ void async_load_b128(unsigned lds_off, const void* gptr) {
  asm volatile("global_load_async_to_lds_b128 %0, %1, off"
               :: "v"(lds_off), "v"(gptr) : "memory");
}

template <int N>
__device__ __forceinline__ void wait_async() {
  asm volatile("s_wait_asynccnt %0" :: "n"(N) : "memory");
}

// TDM: load a contiguous 4KB tile (2048 x bf16, 1 row) from global to LDS.
// D# packed per CDNA5 ISA 8.3/8.4: group0 {count=1, lds_addr, global_addr,
// type=2}; group1 {data_size=2B, tensor_dim0=2048, tensor_dim1=1,
// tile_dim0=2048, tile_dim1=1, dim0_stride=2048}.  6-arg builtin form.
__device__ __forceinline__ void tdm_load_4kb(const void* gsrc, unsigned lds_off) {
  unsigned long long ga = (unsigned long long)(size_t)gsrc;
  v4u g0 = { 1u, lds_off, (unsigned)ga,
             (unsigned)(((ga >> 32) & 0x1FFFFFFull) | (2u << 30)) };
  v8i g1 = { (int)(1u << 16),            // data_size = 2 bytes
             (int)(2048u << 16),         // tensor_dim0[15:0] in bits[63:48]
             (int)(1u << 16),            // tensor_dim1[15:0] in bits[95:80]
             (int)(2048u << 16),         // tile_dim0 in bits[127:112]
             1,                          // tile_dim1 = 1
             2048,                       // tensor_dim0_stride[31:0]
             0, 0 };
  v4i z4 = { 0, 0, 0, 0 };
  v8i z8 = { 0, 0, 0, 0, 0, 0, 0, 0 };
  __builtin_amdgcn_tensor_load_to_lds(g0, g1, z4, z4, z8, 0);
}

__device__ __forceinline__ unsigned lds_off_of(const void* p) {
  return (unsigned)(size_t)p;   // LDS aperture keeps offset in low 32 bits
}

// Load a 16x32 bf16 A/B fragment from an LDS tile stored row-major
// (rows = M for A, rows = N for B i.e. B^T storage), per CDNA5 ISA layout:
// lanes 0-15: row = lane,    K = {kbase+0..7,  kbase+16..23}
// lanes16-31: row = lane-16, K = {kbase+8..15, kbase+24..31}
__device__ __forceinline__ v16bf load_frag(const unsigned short* p0, int stride,
                                           int lane, int kbase) {
  const int r    = lane & 15;
  const int half = (lane >> 4) & 1;
  const unsigned short* p = p0 + r * stride + kbase + half * 8;
  FragU f;
  f.u[0] = *(const uint4*)(p);
  f.u[1] = *(const uint4*)(p + 16);
  return f.v;
}

__device__ __forceinline__ v8f wmma_bf16(v16bf a, v16bf b, v8f c) {
  return __builtin_amdgcn_wmma_f32_16x16x32_bf16(false, a, false, b,
                                                 (short)0, c, false, false);
}

// ---------------- pack kernels ----------------

// window-partition gather + fp32->bf16: xw[bw][t][c], bw=(b*8+wy)*8+wx, t=ty*8+tx
__global__ void pack_x_kernel(const float* __restrict__ x,
                              unsigned short* __restrict__ xw) {
  int gid = blockIdx.x * blockDim.x + threadIdx.x;   // 0 .. 4194303 (4 ch each)
  int c4 = gid & 63; int rest = gid >> 6;
  int t = rest & 63; int bw = rest >> 6;
  int b = bw >> 6;   int w = bw & 63;
  int wy = w >> 3, wx = w & 7, ty = t >> 3, tx = t & 7;
  int n = (wy * 8 + ty) * 64 + wx * 8 + tx;
  const float4 s = *(const float4*)(x + ((size_t)b * 4096 + n) * 256 + c4 * 4);
  ushort4 o;
  o.x = f2bf(s.x); o.y = f2bf(s.y); o.z = f2bf(s.z); o.w = f2bf(s.w);
  *(ushort4*)(xw + (size_t)gid * 4) = o;
}

// transpose + convert all small operands so GEMM B operands are B^T row-major
__global__ void pack_small_kernel(const float* __restrict__ Wqkv,
                                  const float* __restrict__ Wproj,
                                  const float* __restrict__ Ek,
                                  const float* __restrict__ Ev,
                                  const float* __restrict__ kbank,
                                  const float* __restrict__ vbank,
                                  unsigned short* __restrict__ wqkvT,
                                  unsigned short* __restrict__ wprojT,
                                  unsigned short* __restrict__ ekT,
                                  unsigned short* __restrict__ evT,
                                  unsigned short* __restrict__ kbT,
                                  unsigned short* __restrict__ vbT) {
  int gid = blockIdx.x * blockDim.x + threadIdx.x;
  if (gid < 196608) {  // W_qkv^T [768][256]
    int j = gid >> 8; int c = gid & 255;
    wqkvT[gid] = f2bf(Wqkv[c * 768 + j]); return;
  }
  gid -= 196608;
  if (gid < 65536) {   // W_proj^T [256][256]
    int j = gid >> 8; int c = gid & 255;
    wprojT[gid] = f2bf(Wproj[c * 256 + j]); return;
  }
  gid -= 65536;
  if (gid < 2048) {    // E_k^T [32][64]
    int kk = gid >> 6; int n = gid & 63;
    ekT[gid] = f2bf(Ek[n * 32 + kk]); return;
  }
  gid -= 2048;
  if (gid < 2048) {    // E_v^T [32][64]
    int kk = gid >> 6; int n = gid & 63;
    evT[gid] = f2bf(Ev[n * 32 + kk]); return;
  }
  gid -= 2048;
  if (gid < 4096) {    // k_bank per head [h][g][d]
    int h = gid >> 9; int g = (gid >> 5) & 15; int d = gid & 31;
    kbT[gid] = f2bf(kbank[g * 256 + h * 32 + d]); return;
  }
  gid -= 4096;
  if (gid < 4096) {    // v_bank per head [h][g][d]
    int h = gid >> 9; int g = (gid >> 5) & 15; int d = gid & 31;
    vbT[gid] = f2bf(vbank[g * 256 + h * 32 + d]); return;
  }
}

// ---------------- QKV projection GEMM: [65536x256] @ [256x768] ----------------
// Double-buffered LDS tiles fed by async global->LDS; WMMAs of step i overlap
// the ASYNCcnt-tracked loads of step i+1.
__global__ __launch_bounds__(128)
void qkv_gemm_kernel(const unsigned short* __restrict__ xw,
                     const unsigned short* __restrict__ wT,   // [768][256]
                     const float* __restrict__ bias,
                     unsigned short* __restrict__ qout,
                     unsigned short* __restrict__ kout,
                     unsigned short* __restrict__ vout) {
  __shared__ __align__(16) unsigned short sA[2][64 * 32];
  __shared__ __align__(16) unsigned short sB[2][64 * 32];
  const int tid = threadIdx.x, lane = tid & 31, wv = tid >> 5;
  const int m0 = blockIdx.x * 64;
  const int n0 = blockIdx.y * 64;
  v8f acc[4] = {};
  const unsigned short* Abase = xw + (size_t)m0 * 256;
  const unsigned short* Bbase = wT + (size_t)n0 * 256;
  const unsigned aOff0 = lds_off_of(&sA[0][0]);
  const unsigned bOff0 = lds_off_of(&sB[0][0]);

  auto issue_tile = [&](int ks, int buf) {
    const int k0 = ks * 32;
    #pragma unroll
    for (int s = 0; s < 2; ++s) {
      int vv = tid + 128 * s;                 // 256 x 16B vectors per tile
      int row = vv >> 2, kk = (vv & 3) * 8;
      async_load_b128(aOff0 + (unsigned)(buf * 4096 + (row * 32 + kk) * 2),
                      Abase + (size_t)row * 256 + k0 + kk);
      async_load_b128(bOff0 + (unsigned)(buf * 4096 + (row * 32 + kk) * 2),
                      Bbase + (size_t)row * 256 + k0 + kk);
    }
  };

  issue_tile(0, 0);
  for (int ks = 0; ks < 8; ++ks) {
    if (ks < 7) {
      issue_tile(ks + 1, (ks + 1) & 1);       // 4 async instrs/wave in flight
      __builtin_prefetch(Abase + (size_t)(tid >> 1) * 256 + ks * 32 + 64, 0, 1);
      wait_async<4>();                         // tile ks complete (in-order)
    } else {
      wait_async<0>();
    }
    __syncthreads();                           // cross-wave visibility
    const unsigned short* a_ = sA[ks & 1];
    const unsigned short* b_ = sB[ks & 1];
    v16bf a = load_frag(a_ + wv * 16 * 32, 32, lane, 0);
    #pragma unroll
    for (int nt = 0; nt < 4; ++nt) {
      v16bf b = load_frag(b_ + nt * 16 * 32, 32, lane, 0);
      acc[nt] = wmma_bf16(a, b, acc[nt]);
    }
    __syncthreads();                           // done reading before overwrite
  }

  const int nloc = lane & 15, half = lane >> 4;
  #pragma unroll
  for (int nt = 0; nt < 4; ++nt) {
    const int j = n0 + nt * 16 + nloc;        // 0..767
    const float bb = bias[j];
    const int part = j >> 8;                  // 0=q 1=k 2=v
    const int hh = (j >> 5) & 7;
    const int dd = j & 31;
    unsigned short* dst = (part == 0) ? qout : (part == 1) ? kout : vout;
    #pragma unroll
    for (int r = 0; r < 8; ++r) {
      const int tok = m0 + wv * 16 + half * 8 + r;
      const int bw = tok >> 6, t = tok & 63;
      dst[(((size_t)(bw * 8 + hh)) * 64 + t) * 32 + dd] = f2bf(acc[nt][r] + bb);
    }
  }
}

// ------- fused per-(window,head) attention: compress + bank + SDPA -------
__global__ __launch_bounds__(128)
void attn_kernel(const unsigned short* __restrict__ q,
                 const unsigned short* __restrict__ k,
                 const unsigned short* __restrict__ v,
                 const unsigned short* __restrict__ ekT,
                 const unsigned short* __restrict__ evT,
                 const unsigned short* __restrict__ kbT,
                 const unsigned short* __restrict__ vbT,
                 unsigned short* __restrict__ aout) {
  __shared__ __align__(16) unsigned short sm[17920];
  __shared__ float sS[64 * 48];
  unsigned short* sQ   = sm;               // [64 q ][32 d]
  unsigned short* sKt  = sm + 2048;        // [32 d ][64 n]  (K^T: B^T for compression)
  unsigned short* sVt  = sm + 4096;        // [32 d ][64 n]  (V^T)
  unsigned short* sEk  = sm + 6144;        // [32 lk][64 n]  (TDM-loaded)
  unsigned short* sEv  = sm + 8192;        // [32 lk][64 n]  (TDM-loaded)
  unsigned short* sKf  = sm + 10240;       // [48 j ][32 d]  K_full (B^T for scores)
  unsigned short* sVfT = sm + 11776;       // [32 d ][64 j]  V_full^T, j>=48 zero pad
  unsigned short* sP   = sm + 13824;       // [64 q ][64 j]  staging now, probs later
  const int tid = threadIdx.x, lane = tid & 31, wv = tid >> 5;
  const int h  = blockIdx.x;
  const int bw = blockIdx.y;
  const size_t base = ((size_t)(bw * 8 + h)) * 64 * 32;

  // ---- TDM pulls the uniform E tiles (one wave issues; TENSORcnt sync) ----
  if (tid < 32) {
    tdm_load_4kb(ekT, lds_off_of(sEk));
    tdm_load_4kb(evT, lds_off_of(sEv));
  }

  // ---- async global->LDS for Q and row-major K/V staging (in sP area) ----
  const unsigned qOff = lds_off_of(sQ);
  const unsigned gOff = lds_off_of(sP);    // staging: K at +0, V at +4096 bytes
  #pragma unroll
  for (int s = 0; s < 2; ++s) {
    int e = tid + 128 * s;                 // 256 x 16B vectors each
    async_load_b128(qOff + (unsigned)e * 16, (const char*)(q + base) + e * 16);
    async_load_b128(gOff + (unsigned)e * 16, (const char*)(k + base) + e * 16);
    async_load_b128(gOff + 4096u + (unsigned)e * 16,
                    (const char*)(v + base) + e * 16);
  }

  // ---- global bank rows + zero padding (tiny, synchronous) ----
  #pragma unroll
  for (int s = 0; s < 4; ++s) {
    int e = tid + 128 * s;                 // 0..511
    int g = e >> 5, d = e & 31;
    sKf[(32 + g) * 32 + d] = kbT[h * 512 + e];
    sVfT[d * 64 + 32 + g]  = vbT[h * 512 + e];
  }
  #pragma unroll
  for (int s = 0; s < 4; ++s) {
    int e = tid + 128 * s;
    int d = e >> 4, j = e & 15;
    sVfT[d * 64 + 48 + j] = 0;
  }

  wait_async<0>();
  __builtin_amdgcn_s_wait_tensorcnt(0);
  __syncthreads();

  // ---- LDS->LDS transpose of K,V staging into B^T layout ----
  #pragma unroll
  for (int s = 0; s < 16; ++s) {
    int e = tid * 16 + s;                  // 0..2047
    int n = e >> 5, d = e & 31;
    sKt[d * 64 + n] = sP[e];
    sVt[d * 64 + n] = sP[2048 + e];
  }
  __syncthreads();

  // Linformer compression: k_c = E^T(32x64) @ K(64x32), same for V
  {
    const int mt = wv >> 1, nt = wv & 1;   // 2x2 output tiles, one per wave
    v8f ck = {}, cv = {};
    #pragma unroll
    for (int ks = 0; ks < 2; ++ks) {
      v16bf ea = load_frag(sEk + mt * 16 * 64, 64, lane, ks * 32);
      v16bf kb = load_frag(sKt + nt * 16 * 64, 64, lane, ks * 32);
      ck = wmma_bf16(ea, kb, ck);
      v16bf eb = load_frag(sEv + mt * 16 * 64, 64, lane, ks * 32);
      v16bf vb = load_frag(sVt + nt * 16 * 64, 64, lane, ks * 32);
      cv = wmma_bf16(eb, vb, cv);
    }
    const int n = lane & 15, half = lane >> 4;
    #pragma unroll
    for (int r = 0; r < 8; ++r) {
      const int m = mt * 16 + half * 8 + r;   // lk row
      const int d = nt * 16 + n;
      sKf[m * 32 + d]  = f2bf(ck[r]);
      sVfT[d * 64 + m] = f2bf(cv[r]);
    }
  }
  __syncthreads();

  // scores = Q(64x32) @ K_full^T -> [64 q][48 j]
  {
    const float scale = 0.17677669529663687f;  // 1/sqrt(32)
    const int mt = wv;
    v16bf qa = load_frag(sQ + mt * 16 * 32, 32, lane, 0);
    const int n = lane & 15, half = lane >> 4;
    #pragma unroll
    for (int nt = 0; nt < 3; ++nt) {
      v8f sc = {};
      v16bf kf = load_frag(sKf + nt * 16 * 32, 32, lane, 0);
      sc = wmma_bf16(qa, kf, sc);
      #pragma unroll
      for (int r = 0; r < 8; ++r) {
        const int m = mt * 16 + half * 8 + r;
        sS[m * 48 + nt * 16 + n] = sc[r] * scale;
      }
    }
  }
  __syncthreads();

  // row softmax (48 wide), write bf16 probs padded to K=64
  if (tid < 64) {
    float mx = -3.4e38f;
    #pragma unroll
    for (int j = 0; j < 48; ++j) mx = fmaxf(mx, sS[tid * 48 + j]);
    float sum = 0.f;
    #pragma unroll
    for (int j = 0; j < 48; ++j) {
      float e = __expf(sS[tid * 48 + j] - mx);
      sum += e; sS[tid * 48 + j] = e;
    }
    const float inv = 1.f / sum;
    #pragma unroll
    for (int j = 0; j < 48; ++j) sP[tid * 64 + j] = f2bf(sS[tid * 48 + j] * inv);
    #pragma unroll
    for (int j = 48; j < 64; ++j) sP[tid * 64 + j] = 0;
  }
  __syncthreads();

  // out = P(64x64pad) @ V_full(64pad x 32), store merged-head bf16
  {
    const int mt = wv;
    v8f o0 = {}, o1 = {};
    #pragma unroll
    for (int ks = 0; ks < 2; ++ks) {
      v16bf pa = load_frag(sP + mt * 16 * 64, 64, lane, ks * 32);
      v16bf b0 = load_frag(sVfT, 64, lane, ks * 32);
      o0 = wmma_bf16(pa, b0, o0);
      v16bf b1 = load_frag(sVfT + 16 * 64, 64, lane, ks * 32);
      o1 = wmma_bf16(pa, b1, o1);
    }
    const int n = lane & 15, half = lane >> 4;
    #pragma unroll
    for (int r = 0; r < 8; ++r) {
      const int m = mt * 16 + half * 8 + r;
      const size_t row = (size_t)bw * 64 + m;
      aout[row * 256 + h * 32 + n]      = f2bf(o0[r]);
      aout[row * 256 + h * 32 + 16 + n] = f2bf(o1[r]);
    }
  }
}

// -------- projection GEMM + fused window-reverse scatter (fp32 out) --------
__global__ __launch_bounds__(128)
void proj_gemm_kernel(const unsigned short* __restrict__ A,   // [65536][256] bf16
                      const unsigned short* __restrict__ wT,  // [256][256] bf16
                      const float* __restrict__ bias,
                      float* __restrict__ out) {
  __shared__ __align__(16) unsigned short sA[2][64 * 32];
  __shared__ __align__(16) unsigned short sB[2][64 * 32];
  const int tid = threadIdx.x, lane = tid & 31, wv = tid >> 5;
  const int m0 = blockIdx.x * 64;
  const int n0 = blockIdx.y * 64;
  v8f acc[4] = {};
  const unsigned short* Abase = A + (size_t)m0 * 256;
  const unsigned short* Bbase = wT + (size_t)n0 * 256;
  const unsigned aOff0 = lds_off_of(&sA[0][0]);
  const unsigned bOff0 = lds_off_of(&sB[0][0]);

  auto issue_tile = [&](int ks, int buf) {
    const int k0 = ks * 32;
    #pragma unroll
    for (int s = 0; s < 2; ++s) {
      int vv = tid + 128 * s;
      int row = vv >> 2, kk = (vv & 3) * 8;
      async_load_b128(aOff0 + (unsigned)(buf * 4096 + (row * 32 + kk) * 2),
                      Abase + (size_t)row * 256 + k0 + kk);
      async_load_b128(bOff0 + (unsigned)(buf * 4096 + (row * 32 + kk) * 2),
                      Bbase + (size_t)row * 256 + k0 + kk);
    }
  };

  issue_tile(0, 0);
  for (int ks = 0; ks < 8; ++ks) {
    if (ks < 7) {
      issue_tile(ks + 1, (ks + 1) & 1);
      __builtin_prefetch(Abase + (size_t)(tid >> 1) * 256 + ks * 32 + 64, 0, 1);
      wait_async<4>();
    } else {
      wait_async<0>();
    }
    __syncthreads();
    const unsigned short* a_ = sA[ks & 1];
    const unsigned short* b_ = sB[ks & 1];
    v16bf a = load_frag(a_ + wv * 16 * 32, 32, lane, 0);
    #pragma unroll
    for (int nt = 0; nt < 4; ++nt) {
      v16bf b = load_frag(b_ + nt * 16 * 32, 32, lane, 0);
      acc[nt] = wmma_bf16(a, b, acc[nt]);
    }
    __syncthreads();
  }

  const int nloc = lane & 15, half = lane >> 4;
  #pragma unroll
  for (int nt = 0; nt < 4; ++nt) {
    const int col = n0 + nt * 16 + nloc;
    const float bb = bias[col];
    #pragma unroll
    for (int r = 0; r < 8; ++r) {
      const int tok = m0 + wv * 16 + half * 8 + r;
      const int bw = tok >> 6, t = tok & 63;
      const int b  = bw >> 6,  w = bw & 63;
      const int wy = w >> 3, wx = w & 7, ty = t >> 3, tx = t & 7;
      const int n  = (wy * 8 + ty) * 64 + wx * 8 + tx;   // window reverse
      out[((size_t)b * 4096 + n) * 256 + col] = acc[nt][r] + bb;
    }
  }
}

// ---------------- host launcher ----------------
extern "C" void kernel_launch(void* const* d_in, const int* in_sizes, int n_in,
                              void* d_out, int out_size, void* d_ws, size_t ws_size,
                              hipStream_t stream) {
  const float* x     = (const float*)d_in[0];
  const float* Wqkv  = (const float*)d_in[1];
  const float* bqkv  = (const float*)d_in[2];
  const float* Ek    = (const float*)d_in[3];
  const float* Ev    = (const float*)d_in[4];
  const float* kbank = (const float*)d_in[5];
  const float* vbank = (const float*)d_in[6];
  const float* Wproj = (const float*)d_in[7];
  const float* bproj = (const float*)d_in[8];
  // d_in[9]=window_size, d_in[10]=num_heads (compile-time constants here)

  char* ws = (char*)d_ws;
  const size_t SZ_XW = 33554432;               // 65536*256*2 (reused as attn out)
  unsigned short* xw     = (unsigned short*)(ws);
  unsigned short* qbuf   = (unsigned short*)(ws + SZ_XW);
  unsigned short* kbuf   = (unsigned short*)(ws + SZ_XW * 2);
  unsigned short* vbuf   = (unsigned short*)(ws + SZ_XW * 3);
  unsigned short* wqkvT  = (unsigned short*)(ws + SZ_XW * 4);
  unsigned short* wprojT = (unsigned short*)(ws + SZ_XW * 4 + 393216);
  unsigned short* ekT    = (unsigned short*)(ws + SZ_XW * 4 + 393216 + 131072);
  unsigned short* evT    = ekT + 2048;
  unsigned short* kbT    = evT + 2048;
  unsigned short* vbT    = kbT + 4096;

  pack_x_kernel<<<16384, 256, 0, stream>>>(x, xw);
  pack_small_kernel<<<(274432 + 255) / 256, 256, 0, stream>>>(
      Wqkv, Wproj, Ek, Ev, kbank, vbank, wqkvT, wprojT, ekT, evT, kbT, vbT);
  qkv_gemm_kernel<<<dim3(1024, 12), 128, 0, stream>>>(xw, wqkvT, bqkv,
                                                      qbuf, kbuf, vbuf);
  attn_kernel<<<dim3(8, 1024), 128, 0, stream>>>(qbuf, kbuf, vbuf,
                                                 ekT, evT, kbT, vbT, xw);
  proj_gemm_kernel<<<dim3(1024, 4), 128, 0, stream>>>(xw, wprojT, bproj,
                                                      (float*)d_out);
}
